// FrameRNN_39926015984305
// MI455X (gfx1250) — matrix operand order repaired and checked by reading, not compiled, not run on hardware
//
#include <hip/hip_runtime.h>
#include <hip/hip_bf16.h>
#include <stdint.h>

// ---------------- CDNA5 WMMA types ----------------
typedef __attribute__((ext_vector_type(16))) __bf16 v16bf;
typedef __attribute__((ext_vector_type(8)))  __bf16 v8bf;
typedef __attribute__((ext_vector_type(8)))  float  v8f;

#define B_   64
#define T_   64
#define FS_  16
#define DIM_ 1024
#define G3_  3072    // 3*DIM
#define NP_  16384   // DIM*FS

// ---- CDNA5 async global->LDS path (guarded; falls back to manual) ----
#if __has_builtin(__builtin_amdgcn_global_load_async_to_lds_b128) && \
    __has_builtin(__builtin_amdgcn_s_wait_asynccnt)
#define ASYNC_LDS 1
// exact parameter types per hipcc diagnostic: b128 payload is a 16-byte int
// vector; src in AS(1) (global), dst in AS(3) (LDS).
typedef int v4i_ __attribute__((__vector_size__(16)));
#define ASYNC_GPTR(p) ((__attribute__((address_space(1))) v4i_*)(p))
#define ASYNC_LPTR(p) ((__attribute__((address_space(3))) v4i_*)(p))
#else
#define ASYNC_LDS 0
#endif

// ---- fast math (native TRANS ops; co-execute with WMMA per ISA 8.4) ----
__device__ __forceinline__ float fast_rcp(float x) {
#if __has_builtin(__builtin_amdgcn_rcpf)
  return __builtin_amdgcn_rcpf(x);
#else
  return 1.0f / x;
#endif
}
__device__ __forceinline__ float fast_tanh(float x) {
#if __has_builtin(__builtin_amdgcn_tanhf)
  return __builtin_amdgcn_tanhf(x);
#else
  return tanhf(x);
#endif
}
__device__ __forceinline__ float sigmoidf_(float x) {
  return fast_rcp(1.0f + __expf(-x));
}

#define WMMA_BF16(A, Bv, Cv) \
  __builtin_amdgcn_wmma_f32_16x16x32_bf16(false, (A), false, (Bv), (short)0, (Cv), false, false)

// =====================================================================
// Pack f32 weight matrix [K,N] row-major -> bf16 WMMA-B fragment layout
// Bp[k/32][n][k%32]: per K-chunk each lane does one contiguous 32B load,
// and a 256-column span of one chunk is a contiguous 16KB region.
// =====================================================================
__global__ void pack_b(const float* __restrict__ W, __bf16* __restrict__ Bp,
                       int K, int N) {
  long long i = (long long)blockIdx.x * blockDim.x + threadIdx.x;
  long long total = (long long)K * N;
  if (i >= total) return;
  int k = (int)(i / N);
  int n = (int)(i % N);
  Bp[(((long long)(k >> 5) * N + n) << 5) + (k & 31)] = (__bf16)W[i];
}

// =====================================================================
// cell_in[t*B+b, d] = frames(b,t,:) @ Wc[:,d] + big(b,t,d), stored bf16
// =====================================================================
__global__ void build_xs(const int* __restrict__ seq,   // [B, T*FS]
                         const float* __restrict__ Wc,  // [FS, DIM]
                         const float* __restrict__ big, // [B, T, DIM]
                         __bf16* __restrict__ xsb) {    // [T*B, DIM]
  int i = blockIdx.x * blockDim.x + threadIdx.x;
  if (i >= T_ * B_ * DIM_) return;
  int d  = i & (DIM_ - 1);
  int tb = i >> 10;
  int t = tb >> 6;
  int b = tb & 63;
  float acc = big[((long long)b * T_ + t) * DIM_ + d];
  const int* srow = seq + (long long)b * (T_ * FS_) + t * FS_;
#pragma unroll
  for (int f = 0; f < FS_; ++f) {
    float v = (float)srow[f] * (1.0f / 256.0f) - 2.0f;
    acc += v * Wc[f * DIM_ + d];
  }
  xsb[(long long)tb * DIM_ + d] = (__bf16)acc;
}

__global__ void init_h(const float* __restrict__ fs,
                       float* __restrict__ h0f, float* __restrict__ h1f,
                       __bf16* __restrict__ h0b, __bf16* __restrict__ h1b) {
  int i = blockIdx.x * blockDim.x + threadIdx.x;
  if (i >= B_ * DIM_) return;
  float a = fs[i];
  float c = fs[B_ * DIM_ + i];
  h0f[i] = a; h1f[i] = c;
  h0b[i] = (__bf16)a; h1b[i] = (__bf16)c;
}

// =====================================================================
// LDS-staged, double-buffered WMMA GEMM; block = 256 threads (8 waves as
// 2x4) -> 64x256 macro-tile, each wave a 32x64 register block (8 WMMAs
// per K-chunk).  Staging uses GLOBAL_LOAD_ASYNC_TO_LDS (ASYNCcnt) when
// the builtin exists; otherwise a register-staged double buffer.  LDS
// rows padded to 40 bf16 (20 dwords) -> conflict-free ds_load_b128.
// =====================================================================
#define APAD 40   // 32 + 8 bf16 row pitch

template <bool PROJ>
__global__ void gemm_bf16_lds(const __bf16* __restrict__ A,
                              const __bf16* __restrict__ Bp,
                              const float*  __restrict__ bias,
                              float* __restrict__ C,
                              int M, int N, int K) {
  __shared__ __bf16 sA[2][64 * APAD];    // 2 x 5 KB
  __shared__ __bf16 sB[2][256 * APAD];   // 2 x 20 KB

  int tid  = threadIdx.x;             // 0..255
  int wq   = tid >> 5;                // wave in block, 0..7
  int lane = tid & 31;
  int half = lane >> 4, l15 = lane & 15;
  int wm = wq >> 2, wn = wq & 3;      // 2 x 4 wave grid

  int nbn = N >> 8;                   // N / 256
  int gbm = blockIdx.x / nbn;
  int gbn = blockIdx.x % nbn;
  int row0 = gbm * 64;
  int col0 = gbn * 256;

  // staging: A row (tid>>2), 16B segment (tid&3); B column tid (32 el)
  const int aoff = (tid >> 2) * APAD + ((tid & 3) << 3);
  const int boff = tid * APAD;
  const __bf16* gA = A + (long long)(row0 + (tid >> 2)) * K + ((tid & 3) << 3);
  const __bf16* gB = Bp + ((long long)col0 << 5) + (tid << 5);
  const long long bchunk = (long long)N << 5;

  // fragment read offsets (LDS)
  const int aread = (wm * 32 + l15) * APAD + half * 8;
  const int bread = (wn * 64 + l15) * APAD + (half << 3);

  const int NC = K >> 5;
  v8f acc[2][4] = {};

  auto compute = [&](int buf) {
    const __bf16* aF = &sA[buf][aread];
    const __bf16* bF = &sB[buf][bread];
    union { v16bf v; v8bf h[2]; } a0, a1;
    a0.h[0] = *(const v8bf*)(aF);
    a0.h[1] = *(const v8bf*)(aF + 16);
    a1.h[0] = *(const v8bf*)(aF + 16 * APAD);
    a1.h[1] = *(const v8bf*)(aF + 16 * APAD + 16);
#pragma unroll
    for (int j = 0; j < 4; ++j) {
      union { v16bf v; v8bf h[2]; } bb;
      bb.h[0] = *(const v8bf*)(bF + j * 16 * APAD);
      bb.h[1] = *(const v8bf*)(bF + j * 16 * APAD + 16);
      acc[0][j] = WMMA_BF16(a0.v, bb.v, acc[0][j]);
      acc[1][j] = WMMA_BF16(a1.v, bb.v, acc[1][j]);
    }
  };

#if ASYNC_LDS
  auto issue = [&](int c, int buf) {
    const __bf16* gAc = gA + (c << 5);
    const __bf16* gBc = gB + (long long)c * bchunk;
    __builtin_amdgcn_global_load_async_to_lds_b128(
        ASYNC_GPTR(gAc), ASYNC_LPTR(&sA[buf][aoff]), 0, 0);
#pragma unroll
    for (int u = 0; u < 4; ++u)
      __builtin_amdgcn_global_load_async_to_lds_b128(
          ASYNC_GPTR(gBc + u * 8), ASYNC_LPTR(&sB[buf][boff + u * 8]), 0, 0);
  };
  issue(0, 0);
  __builtin_amdgcn_s_wait_asynccnt(0);
  __syncthreads();
  for (int c = 0; c < NC; ++c) {
    int cur = c & 1;
    if (c + 1 < NC) issue(c + 1, cur ^ 1);
    compute(cur);
    __builtin_amdgcn_s_wait_asynccnt(0);
    __syncthreads();
  }
#else
  // manual double buffer: global->VGPR early, VGPR->LDS after compute
  {
    *(v8bf*)(&sA[0][aoff]) = *(const v8bf*)(gA);
#pragma unroll
    for (int u = 0; u < 4; ++u)
      *(v8bf*)(&sB[0][boff + u * 8]) = *(const v8bf*)(gB + u * 8);
  }
  __syncthreads();
  for (int c = 0; c < NC; ++c) {
    int cur = c & 1;
    v8bf ra = {};
    v8bf rb[4] = {};
    if (c + 1 < NC) {
      ra = *(const v8bf*)(gA + ((c + 1) << 5));
      const __bf16* gBc = gB + (long long)(c + 1) * bchunk;
#pragma unroll
      for (int u = 0; u < 4; ++u) rb[u] = *(const v8bf*)(gBc + u * 8);
    }
    compute(cur);
    if (c + 1 < NC) {
      *(v8bf*)(&sA[cur ^ 1][aoff]) = ra;
#pragma unroll
      for (int u = 0; u < 4; ++u)
        *(v8bf*)(&sB[cur ^ 1][boff + u * 8]) = rb[u];
    }
    __syncthreads();
  }
#endif

#pragma unroll
  for (int i = 0; i < 2; ++i) {
    int m0 = row0 + wm * 32 + i * 16 + half * 8;
#pragma unroll
    for (int j = 0; j < 4; ++j) {
      int n = col0 + wn * 64 + j * 16 + l15;
      if (PROJ) {
        int s = n >> 10, d = n & (DIM_ - 1);
#pragma unroll
        for (int r = 0; r < 8; ++r) {
          int m = m0 + r;            // m = t*B + b
          int t = m >> 6, b = m & 63;
          C[((long long)b * (T_ * FS_) + t * FS_ + s) * DIM_ + d] = acc[i][j][r];
        }
      } else {
        float bb = bias ? bias[n] : 0.0f;
#pragma unroll
        for (int r = 0; r < 8; ++r)
          C[(long long)(m0 + r) * N + n] = acc[i][j][r] + bb;
      }
    }
  }
}

// =====================================================================
// Fused recurrence step, layer 0.  All 8 waves of a block share the same
// 16-row A tile (mi = blockIdx/8), staged once into LDS (rows padded to
// 1032 bf16 -> fragment lanes on banks 4*l15, distinct).
// =====================================================================
#define HPITCH 1032   // 1024 + 8 bf16

__global__ void step_layer0(const __bf16* __restrict__ hA,   // h0 bf16 (old)
                            const __bf16* __restrict__ Bp,   // rk0 packed [*,G3]
                            const float*  __restrict__ gi,   // gi0 + t*B*G3
                            const float*  __restrict__ bias, // b0[1] [G3]
                            float* __restrict__ hf,          // h0 f32 in/out
                            __bf16* __restrict__ hOut) {     // h0 bf16 (new)
  __shared__ __bf16 sH[16 * HPITCH];   // ~32 KB

  int tid  = threadIdx.x;
  int wq   = tid >> 5;
  int lane = tid & 31;
  int half = lane >> 4, l15 = lane & 15;
  int mi = blockIdx.x >> 3;                      // 0..3 (8 blocks per mi)
  int dj = ((blockIdx.x & 7) << 3) + wq;         // 0..63

  // stage A tile: thread t copies row t>>4, 64-el segment t&15 (8 x b128)
  {
    int rr = tid >> 4, seg = tid & 15;
    const __bf16* src = hA + (((long long)(mi * 16 + rr)) << 10) + seg * 64;
    __bf16* dst = sH + rr * HPITCH + seg * 64;
#pragma unroll
    for (int u = 0; u < 8; ++u)
      *(v8bf*)(dst + u * 8) = *(const v8bf*)(src + u * 8);
  }
  __syncthreads();

  const __bf16* aL = sH + l15 * HPITCH + half * 16;
  const long long bstride = (long long)G3_ << 5;
  const long long gofs = (long long)DIM_ << 5;
  const __bf16* bz = Bp + ((long long)(dj * 16 + l15) << 5) + (half << 4);

  v8f az = {}, ar = {}, an = {};
#pragma unroll 2
  for (int k0 = 0; k0 < DIM_; k0 += 32) {
    union { v16bf v; v8bf h[2]; } a;
    a.h[0] = *(const v8bf*)(aL + k0);
    a.h[1] = *(const v8bf*)(aL + k0 + 8);
    long long co = (long long)(k0 >> 5) * bstride;
    v16bf vz = *(const v16bf*)(bz + co);
    v16bf vr = *(const v16bf*)(bz + co + gofs);
    v16bf vn = *(const v16bf*)(bz + co + 2 * gofs);
    az = WMMA_BF16(a.v, vz, az);
    ar = WMMA_BF16(a.v, vr, ar);
    an = WMMA_BF16(a.v, vn, an);
  }

  int d = dj * 16 + l15;
  float bzv = bias[d], brv = bias[DIM_ + d], bnv = bias[2 * DIM_ + d];
#pragma unroll
  for (int r = 0; r < 8; ++r) {
    int m = mi * 16 + half * 8 + r;
    const float* gim = gi + (long long)m * G3_;
    float z  = sigmoidf_(gim[d] + az[r] + bzv);
    float rr = sigmoidf_(gim[DIM_ + d] + ar[r] + brv);
    float nn = fast_tanh(gim[2 * DIM_ + d] + rr * (an[r] + bnv));
    long long idx = (long long)m * DIM_ + d;
    float hn = z * hf[idx] + (1.0f - z) * nn;
    hf[idx] = hn;
    hOut[idx] = (__bf16)hn;
  }
}

// =====================================================================
// Fused recurrence step, layer 1: stages both A tiles (h0new, h1old),
// 6 gate tiles / 6 WMMAs per chunk, per-lane combine, writes h1 + outs.
// =====================================================================
__global__ void step_layer1(const __bf16* __restrict__ xA,  // h0 bf16 (new)
                            const __bf16* __restrict__ hA,  // h1 bf16 (old)
                            const __bf16* __restrict__ Kp,  // k1 packed
                            const __bf16* __restrict__ Rp,  // rk1 packed
                            const float*  __restrict__ bi,  // b1[0]
                            const float*  __restrict__ bh,  // b1[1]
                            float* __restrict__ hf,         // h1 f32 in/out
                            __bf16* __restrict__ hOut,      // h1 bf16 (new)
                            __bf16* __restrict__ outs) {    // outsb + t*B*DIM
  __shared__ __bf16 sX[16 * HPITCH];   // ~32 KB
  __shared__ __bf16 sHH[16 * HPITCH];  // ~32 KB

  int tid  = threadIdx.x;
  int wq   = tid >> 5;
  int lane = tid & 31;
  int half = lane >> 4, l15 = lane & 15;
  int mi = blockIdx.x >> 3;
  int dj = ((blockIdx.x & 7) << 3) + wq;

  {
    int rr = tid >> 4, seg = tid & 15;
    long long go = (((long long)(mi * 16 + rr)) << 10) + seg * 64;
    int lo = rr * HPITCH + seg * 64;
#pragma unroll
    for (int u = 0; u < 8; ++u)
      *(v8bf*)(sX + lo + u * 8) = *(const v8bf*)(xA + go + u * 8);
#pragma unroll
    for (int u = 0; u < 8; ++u)
      *(v8bf*)(sHH + lo + u * 8) = *(const v8bf*)(hA + go + u * 8);
  }
  __syncthreads();

  const __bf16* xL = sX  + l15 * HPITCH + half * 16;
  const __bf16* hL = sHH + l15 * HPITCH + half * 16;
  const long long bstride = (long long)G3_ << 5;
  const long long gofs = (long long)DIM_ << 5;
  const long long colo = ((long long)(dj * 16 + l15) << 5) + (half << 4);
  const __bf16* kz = Kp + colo;
  const __bf16* rz = Rp + colo;

  v8f giz = {}, gir = {}, gin = {}, ghz = {}, ghr = {}, ghn = {};
  for (int k0 = 0; k0 < DIM_; k0 += 32) {
    union { v16bf v; v8bf h[2]; } ax, ah;
    ax.h[0] = *(const v8bf*)(xL + k0);
    ax.h[1] = *(const v8bf*)(xL + k0 + 8);
    ah.h[0] = *(const v8bf*)(hL + k0);
    ah.h[1] = *(const v8bf*)(hL + k0 + 8);
    long long co = (long long)(k0 >> 5) * bstride;
    v16bf kvz = *(const v16bf*)(kz + co);
    v16bf kvr = *(const v16bf*)(kz + co + gofs);
    v16bf kvn = *(const v16bf*)(kz + co + 2 * gofs);
    v16bf rvz = *(const v16bf*)(rz + co);
    v16bf rvr = *(const v16bf*)(rz + co + gofs);
    v16bf rvn = *(const v16bf*)(rz + co + 2 * gofs);
    giz = WMMA_BF16(ax.v, kvz, giz);
    gir = WMMA_BF16(ax.v, kvr, gir);
    gin = WMMA_BF16(ax.v, kvn, gin);
    ghz = WMMA_BF16(ah.v, rvz, ghz);
    ghr = WMMA_BF16(ah.v, rvr, ghr);
    ghn = WMMA_BF16(ah.v, rvn, ghn);
  }

  int d = dj * 16 + l15;
  float biz = bi[d], bir = bi[DIM_ + d], bin = bi[2 * DIM_ + d];
  float bhz = bh[d], bhr = bh[DIM_ + d], bhn = bh[2 * DIM_ + d];
#pragma unroll
  for (int r = 0; r < 8; ++r) {
    int m = mi * 16 + half * 8 + r;
    float z  = sigmoidf_(giz[r] + biz + ghz[r] + bhz);
    float rr = sigmoidf_(gir[r] + bir + ghr[r] + bhr);
    float nn = fast_tanh(gin[r] + bin + rr * (ghn[r] + bhn));
    long long idx = (long long)m * DIM_ + d;
    float hn = z * hf[idx] + (1.0f - z) * nn;
    hf[idx] = hn;
    __bf16 hv = (__bf16)hn;
    hOut[idx] = hv;
    outs[idx] = hv;
  }
}

__global__ void copy_final(const float* __restrict__ h0,
                           const float* __restrict__ h1,
                           float* __restrict__ out) {
  int i = blockIdx.x * blockDim.x + threadIdx.x;
  if (i >= B_ * DIM_) return;
  out[i] = h0[i];
  out[B_ * DIM_ + i] = h1[i];
}

// =====================================================================
// Host orchestration
// =====================================================================
extern "C" void kernel_launch(void* const* d_in, const int* in_sizes, int n_in,
                              void* d_out, int out_size, void* d_ws, size_t ws_size,
                              hipStream_t stream) {
  const int*   seq    = (const int*)d_in[0];
  const float* big    = (const float*)d_in[2];
  const float* fstate = (const float*)d_in[3];
  const float* Wc     = (const float*)d_in[4];
  const float* Wp     = (const float*)d_in[5];
  const float* Gk     = (const float*)d_in[6];   // [2, DIM, 3*DIM]
  const float* Grk    = (const float*)d_in[7];   // [2, DIM, 3*DIM]
  const float* Gb     = (const float*)d_in[8];   // [2, 2, 3*DIM]
  float* out = (float*)d_out;

  uintptr_t w = (uintptr_t)d_ws;
  auto carve = [&](size_t bytes) -> uintptr_t {
    uintptr_t p = (w + 255) & ~(uintptr_t)255;
    w = p + bytes;
    return p;
  };
  __bf16* k0p   = (__bf16*)carve((size_t)DIM_ * G3_ * 2);
  __bf16* rk0p  = (__bf16*)carve((size_t)DIM_ * G3_ * 2);
  __bf16* k1p   = (__bf16*)carve((size_t)DIM_ * G3_ * 2);
  __bf16* rk1p  = (__bf16*)carve((size_t)DIM_ * G3_ * 2);
  __bf16* projp = (__bf16*)carve((size_t)DIM_ * NP_ * 2);
  __bf16* xsb   = (__bf16*)carve((size_t)T_ * B_ * DIM_ * 2);
  float*  gi0   = (float*)carve((size_t)T_ * B_ * G3_ * 4);
  float*  h0f   = (float*)carve((size_t)B_ * DIM_ * 4);
  float*  h1f   = (float*)carve((size_t)B_ * DIM_ * 4);
  __bf16* h0bA  = (__bf16*)carve((size_t)B_ * DIM_ * 2);
  __bf16* h0bB  = (__bf16*)carve((size_t)B_ * DIM_ * 2);
  __bf16* h1bA  = (__bf16*)carve((size_t)B_ * DIM_ * 2);
  __bf16* h1bB  = (__bf16*)carve((size_t)B_ * DIM_ * 2);
  __bf16* outsb = (__bf16*)carve((size_t)T_ * B_ * DIM_ * 2);

  // ---- weight repack (f32 -> bf16 WMMA-B layout) ----
  {
    long long n1 = (long long)DIM_ * G3_;
    int blk = (int)((n1 + 255) / 256);
    pack_b<<<blk, 256, 0, stream>>>(Gk,       k0p,  DIM_, G3_);
    pack_b<<<blk, 256, 0, stream>>>(Grk,      rk0p, DIM_, G3_);
    pack_b<<<blk, 256, 0, stream>>>(Gk  + n1, k1p,  DIM_, G3_);
    pack_b<<<blk, 256, 0, stream>>>(Grk + n1, rk1p, DIM_, G3_);
    long long n2 = (long long)DIM_ * NP_;
    pack_b<<<(int)((n2 + 255) / 256), 256, 0, stream>>>(Wp, projp, DIM_, NP_);
  }

  // ---- cell inputs + state init ----
  build_xs<<<(T_ * B_ * DIM_) / 256, 256, 0, stream>>>(seq, Wc, big, xsb);
  init_h<<<(B_ * DIM_) / 256, 256, 0, stream>>>(fstate, h0f, h1f, h0bA, h1bA);

  // ---- time-parallel layer-0 input projection: gi0 = xs @ k0 + b0[0] ----
  {
    int blocks = (T_ * B_ / 64) * (G3_ / 256);   // 64*12 = 768
    gemm_bf16_lds<false><<<blocks, 256, 0, stream>>>(
        xsb, k0p, Gb + 0 * G3_, gi0, T_ * B_, G3_, DIM_);
  }

  // ---- sequential recurrence: 2 fused kernels / step ----
  __bf16 *h0cur = h0bA, *h0nxt = h0bB, *h1cur = h1bA, *h1nxt = h1bB;
  for (int t = 0; t < T_; ++t) {
    step_layer0<<<32, 256, 0, stream>>>(h0cur, rk0p,
                                        gi0 + (long long)t * B_ * G3_,
                                        Gb + 1 * G3_, h0f, h0nxt);
    step_layer1<<<32, 256, 0, stream>>>(h0nxt, h1cur, k1p, rk1p,
                                        Gb + 2 * G3_, Gb + 3 * G3_,
                                        h1f, h1nxt,
                                        outsb + (long long)t * B_ * DIM_);
    __bf16* tmp;
    tmp = h0cur; h0cur = h0nxt; h0nxt = tmp;
    tmp = h1cur; h1cur = h1nxt; h1nxt = tmp;
  }

  // ---- output projection (fused transpose/reshape into store) ----
  {
    int blocks = (T_ * B_ / 64) * (NP_ / 256);   // 64*64 = 4096
    gemm_bf16_lds<true><<<blocks, 256, 0, stream>>>(
        outsb, projp, nullptr, out, T_ * B_, NP_, DIM_);
  }

  // ---- final state ----
  copy_final<<<(B_ * DIM_) / 256, 256, 0, stream>>>(
      h0f, h1f, out + (long long)B_ * T_ * FS_ * DIM_);
}
